// BucketRandomAttentionCausal_2877628089032
// MI455X (gfx1250) — matrix operand order repaired and checked by reading, not compiled, not run on hardware
//
#include <hip/hip_runtime.h>
#include <hip/hip_bf16.h>
#include <cstdint>

typedef __bf16 bf16_t;
typedef __attribute__((ext_vector_type(16))) __bf16 v16bf;
typedef __attribute__((ext_vector_type(8)))  __bf16 v8bf;
typedef __attribute__((ext_vector_type(8)))  float  v8f;

union BF16x16 { v16bf v; v8bf h[2]; };

#define LQ 4096
#define BB 8
#define DD 1024
#define HH 16
#define DH 64
#define SS 128
#define MQ (LQ * BB)   // 32768 rows for q/ctx GEMMs

// ---------------------------------------------------------------------------
// CDNA5 async global->LDS DMA (ASYNCcnt) with fallback to synchronous copy.
// clang prototype: void(v4i addrspace(1)*, v4i addrspace(3)*, Imm i32, Imm i32)
// ---------------------------------------------------------------------------
#if defined(__has_builtin)
#if __has_builtin(__builtin_amdgcn_global_load_async_to_lds_b128)
#define HAVE_ASYNC_LDS 1
#endif
#endif
#ifndef HAVE_ASYNC_LDS
#define HAVE_ASYNC_LDS 0
#endif

typedef int v4i_ __attribute__((vector_size(16)));
typedef __attribute__((address_space(1))) v4i_ glb_v4i;
typedef __attribute__((address_space(3))) v4i_ lds_v4i;

__device__ __forceinline__ void async_or_copy_b128(void* lds_dst, const void* g_src) {
#if HAVE_ASYNC_LDS
    __builtin_amdgcn_global_load_async_to_lds_b128(
        (glb_v4i*)(uintptr_t)g_src,
        (lds_v4i*)(unsigned)(uintptr_t)lds_dst, 0, 0);
#else
    *(v8bf*)lds_dst = *(const v8bf*)g_src;
#endif
}

__device__ __forceinline__ void wait_async0() {
#if HAVE_ASYNC_LDS
#if __has_builtin(__builtin_amdgcn_s_wait_asynccnt)
    __builtin_amdgcn_s_wait_asynccnt(0);
#else
    asm volatile("s_wait_asynccnt 0x0" ::: "memory");
#endif
#endif
}

// ---------------------------------------------------------------------------
// f32 -> bf16 conversion (4 elements / thread)
// ---------------------------------------------------------------------------
__global__ void cvt_f32_bf16_kernel(const float* __restrict__ src,
                                    bf16_t* __restrict__ dst, long long n4) {
    long long i = (long long)blockIdx.x * blockDim.x + threadIdx.x;
    if (i >= n4) return;
    float4 f = ((const float4*)src)[i];
    bf16_t* o = dst + i * 4;
    o[0] = (bf16_t)f.x; o[1] = (bf16_t)f.y;
    o[2] = (bf16_t)f.z; o[3] = (bf16_t)f.w;
}

// ---------------------------------------------------------------------------
// gather sampled rows of k and v, convert to bf16.  grid.x = S*B rows.
// ---------------------------------------------------------------------------
__global__ void gather_cvt_kernel(const float* __restrict__ k,
                                  const float* __restrict__ v,
                                  const int* __restrict__ sampled,
                                  bf16_t* __restrict__ kg,
                                  bf16_t* __restrict__ vg) {
    int row = blockIdx.x;            // s*B + b
    int s = row >> 3;
    int b = row & 7;
    size_t src = ((size_t)sampled[s] * BB + b) * DD;
    size_t dst = (size_t)row * DD;
    int t = threadIdx.x;             // 256 threads * 4 elems = 1024
    float4 fk = ((const float4*)(k + src))[t];
    float4 fv = ((const float4*)(v + src))[t];
    bf16_t* ok = kg + dst + t * 4;
    bf16_t* ov = vg + dst + t * 4;
    ok[0] = (bf16_t)fk.x; ok[1] = (bf16_t)fk.y; ok[2] = (bf16_t)fk.z; ok[3] = (bf16_t)fk.w;
    ov[0] = (bf16_t)fv.x; ov[1] = (bf16_t)fv.y; ov[2] = (bf16_t)fv.z; ov[3] = (bf16_t)fv.w;
}

// ---------------------------------------------------------------------------
// C[M,N] = A[M,K] * W[N,K]^T + bias   (torch linear)
// bf16 in, f32 accumulate via v_wmma_f32_16x16x32_bf16.
// block = 256 thr = 8 waves; tile 128(M) x 128(N); K step 32.
// W tile double-buffered in LDS, staged with async global->LDS DMA.
// Each wave: 16 M-rows x 8 N-tiles -> 8 WMMAs per K step, 1 barrier per step.
// ---------------------------------------------------------------------------
__global__ __launch_bounds__(256)
void gemm_bf16_kernel(const bf16_t* __restrict__ A,
                      const bf16_t* __restrict__ W,
                      const float*  __restrict__ bias,
                      bf16_t* __restrict__ Cb, float* __restrict__ Cf,
                      int M, int N, int K, int store_f32) {
    __shared__ alignas(16) bf16_t ldsW[2][128 * 32];   // 2 x 8 KB
    const int tid   = threadIdx.x;
    const int wave  = tid >> 5;
    const int lane  = tid & 31;
    const int l16   = lane & 15;
    const int lhalf = lane >> 4;
    const int mblk  = blockIdx.x * 128;
    const int nblk  = blockIdx.y * 128;

    v8f acc[8] = {};
    const bf16_t* Arow = A + (size_t)(mblk + wave * 16 + l16) * K;

    // stage a 128x32 W tile: 512 chunks of 8 bf16, 2 per thread
    auto stage = [&](int buf, int kk) {
#pragma unroll
        for (int i = 0; i < 2; ++i) {
            int c  = tid + i * 256;
            int n  = c >> 2;
            int k8 = (c & 3) * 8;
            async_or_copy_b128(&ldsW[buf][n * 32 + k8],
                               &W[(size_t)(nblk + n) * K + kk + k8]);
        }
    };

    stage(0, 0);
    int cur = 0;
    for (int kk = 0; kk < K; kk += 32) {
        wait_async0();           // own DMA done
        __syncthreads();         // everyone's tile[cur] visible; prev reads done
        if (kk + 32 < K) stage(cur ^ 1, kk + 32);   // overlap next tile

        BF16x16 af;              // A frag: lanes 0-15 K{0..15}, 16-31 K{16..31}
        const bf16_t* ap = &Arow[kk + lhalf * 16];
        af.h[0] = *(const v8bf*)&ap[0];
        af.h[1] = *(const v8bf*)&ap[8];

#pragma unroll
        for (int nt = 0; nt < 8; ++nt) {  // B frag: lane half selects K 0..15 / 16..31
            BF16x16 bfm;
            const bf16_t* bp = &ldsW[cur][(nt * 16 + l16) * 32 + lhalf * 16];
            bfm.h[0] = *(const v8bf*)&bp[0];
            bfm.h[1] = *(const v8bf*)&bp[8];
            acc[nt] = __builtin_amdgcn_wmma_f32_16x16x32_bf16(
                false, af.v, false, bfm.v, (short)0, acc[nt], false, false);
        }
        cur ^= 1;
    }

#pragma unroll
    for (int nt = 0; nt < 8; ++nt) {
        int col = nblk + nt * 16 + l16;
        float bv = bias ? bias[col] : 0.0f;
#pragma unroll
        for (int r = 0; r < 8; ++r) {
            int row = mblk + wave * 16 + r + 8 * lhalf;  // D-tile: M = r + 8*(lane>=16)
            float val = acc[nt][r] + bv;
            if (store_f32) Cf[(size_t)row * N + col] = val;
            else           Cb[(size_t)row * N + col] = (bf16_t)val;
        }
    }
}

// ---------------------------------------------------------------------------
// attention: per (b,h) and 64-query strip.
// block = 128 thr = 4 waves, each wave owns 16 query rows.
// scores = Q K^T * 0.125, masked softmax over sorted sampled suffix, ctx = P V.
// ---------------------------------------------------------------------------
__global__ __launch_bounds__(128)
void attn_kernel(const bf16_t* __restrict__ qp, const bf16_t* __restrict__ kp,
                 const bf16_t* __restrict__ vp, const int* __restrict__ sampled,
                 const int* __restrict__ keylen, bf16_t* __restrict__ ctx) {
    __shared__ alignas(16) bf16_t lds_k[SS * DH];      // [s][d]  16 KB
    __shared__ alignas(16) bf16_t lds_vT[DH * SS];     // [d][s]  16 KB
    __shared__ alignas(16) bf16_t lds_p[4][16 * SS];   // per-wave P, 16 KB
    __shared__ int lds_si[SS];
    __shared__ int smkl;

    const int tid   = threadIdx.x;
    const int b     = blockIdx.y >> 4;
    const int hh    = blockIdx.y & 15;
    const int qbase = blockIdx.x * 64;

    if (tid < SS) lds_si[tid] = sampled[tid];
    if (tid == 0) smkl = 0;
    __syncthreads();
    if (tid < SS && lds_si[tid] < keylen[b]) atomicAdd(&smkl, 1);

    // stage K tile [s][d] via async DMA
    for (int c = tid; c < (SS * DH) / 8; c += 128) {
        int s  = c >> 3;
        int d8 = (c & 7) * 8;
        async_or_copy_b128(&lds_k[s * DH + d8],
                           &kp[((size_t)(s * BB + b)) * DD + hh * DH + d8]);
    }
    // stage V tile transposed [d][s] (element-wise, must go through VGPRs)
    for (int c = tid; c < (SS * DH) / 8; c += 128) {
        int s  = c >> 3;
        int d8 = (c & 7) * 8;
        v8bf t = *(const v8bf*)&vp[((size_t)(s * BB + b)) * DD + hh * DH + d8];
#pragma unroll
        for (int j = 0; j < 8; ++j) lds_vT[(d8 + j) * SS + s] = t[j];
    }
    wait_async0();
    __syncthreads();

    const int wave  = tid >> 5;
    const int lane  = tid & 31;
    const int l16   = lane & 15;
    const int lhalf = lane >> 4;

    // ---- scores: 16 queries x 128 samples = 8 WMMA tiles, K = dh = 64 ----
    const bf16_t* Qrow = qp + ((size_t)(qbase + wave * 16 + l16) * BB + b) * DD + hh * DH;
    v8f sc[8] = {};
#pragma unroll
    for (int kk = 0; kk < DH; kk += 32) {
        BF16x16 af;
        const bf16_t* ap = &Qrow[kk + lhalf * 16];
        af.h[0] = *(const v8bf*)&ap[0];
        af.h[1] = *(const v8bf*)&ap[8];
#pragma unroll
        for (int nt = 0; nt < 8; ++nt) {
            BF16x16 bfm;
            const bf16_t* bp = &lds_k[(nt * 16 + l16) * DH + kk + lhalf * 16];
            bfm.h[0] = *(const v8bf*)&bp[0];
            bfm.h[1] = *(const v8bf*)&bp[8];
            sc[nt] = __builtin_amdgcn_wmma_f32_16x16x32_bf16(
                false, af.v, false, bfm.v, (short)0, sc[nt], false, false);
        }
    }

    // ---- masking + softmax (sampled sorted ascending -> suffix mask) ----
    const float scale = 0.125f;   // 1/sqrt(64)
    const int   mkl   = smkl;
    int nv[8];
#pragma unroll
    for (int r = 0; r < 8; ++r) {
        int l = qbase + wave * 16 + r + 8 * lhalf;
        int lo = 0, hi = SS;                   // upper_bound: #samples <= l
        while (lo < hi) { int mid = (lo + hi) >> 1;
                          if (lds_si[mid] <= l) lo = mid + 1; else hi = mid; }
        nv[r] = lo < mkl ? lo : mkl;
    }
    float rmax[8], rsum[8];
#pragma unroll
    for (int r = 0; r < 8; ++r) rmax[r] = -3.0e38f;
#pragma unroll
    for (int nt = 0; nt < 8; ++nt)
#pragma unroll
        for (int r = 0; r < 8; ++r) {
            int sidx = nt * 16 + l16;
            float v = (sidx < nv[r]) ? sc[nt][r] * scale : -3.0e38f;
            sc[nt][r] = v;
            rmax[r] = fmaxf(rmax[r], v);
        }
#pragma unroll
    for (int r = 0; r < 8; ++r)
#pragma unroll
        for (int m = 8; m >= 1; m >>= 1)       // row lives in one 16-lane half
            rmax[r] = fmaxf(rmax[r], __shfl_xor(rmax[r], m, 16));
#pragma unroll
    for (int r = 0; r < 8; ++r) rsum[r] = 0.0f;
#pragma unroll
    for (int nt = 0; nt < 8; ++nt)
#pragma unroll
        for (int r = 0; r < 8; ++r) {
            int sidx = nt * 16 + l16;
            float p = (sidx < nv[r]) ? __expf(sc[nt][r] - rmax[r]) : 0.0f;
            sc[nt][r] = p;
            rsum[r] += p;
        }
#pragma unroll
    for (int r = 0; r < 8; ++r) {
#pragma unroll
        for (int m = 8; m >= 1; m >>= 1)
            rsum[r] += __shfl_xor(rsum[r], m, 16);
        rsum[r] = (rsum[r] > 0.0f) ? 1.0f / rsum[r] : 0.0f;
    }

    // ---- re-fragment P via per-wave LDS (D-layout -> A-layout) ----
    bf16_t* P = lds_p[wave];
#pragma unroll
    for (int nt = 0; nt < 8; ++nt)
#pragma unroll
        for (int r = 0; r < 8; ++r) {
            int m = r + 8 * lhalf;
            P[m * SS + nt * 16 + l16] = (bf16_t)(sc[nt][r] * rsum[r]);
        }

    // ---- ctx = P[16 x 128] * V[128 x 64] : 4 tiles, K steps of 32 ----
    v8f oc[4] = {};
#pragma unroll
    for (int kk = 0; kk < SS; kk += 32) {
        BF16x16 af;
        const bf16_t* ap = &P[l16 * SS + kk + lhalf * 16];
        af.h[0] = *(const v8bf*)&ap[0];
        af.h[1] = *(const v8bf*)&ap[8];
#pragma unroll
        for (int nt = 0; nt < 4; ++nt) {
            BF16x16 bfm;
            const bf16_t* bp = &lds_vT[(nt * 16 + l16) * SS + kk + lhalf * 16];
            bfm.h[0] = *(const v8bf*)&bp[0];
            bfm.h[1] = *(const v8bf*)&bp[8];
            oc[nt] = __builtin_amdgcn_wmma_f32_16x16x32_bf16(
                false, af.v, false, bfm.v, (short)0, oc[nt], false, false);
        }
    }

    // ---- store ctx (bf16, [L*B, D]) ----
#pragma unroll
    for (int nt = 0; nt < 4; ++nt)
#pragma unroll
        for (int r = 0; r < 8; ++r) {
            int m = r + 8 * lhalf;
            int l = qbase + wave * 16 + m;
            int d = nt * 16 + l16;
            ctx[((size_t)l * BB + b) * DD + hh * DH + d] = (bf16_t)oc[nt][r];
        }
}

// ---------------------------------------------------------------------------
extern "C" void kernel_launch(void* const* d_in, const int* in_sizes, int n_in,
                              void* d_out, int out_size, void* d_ws, size_t ws_size,
                              hipStream_t stream) {
    const float* q       = (const float*)d_in[0];
    const float* k       = (const float*)d_in[1];
    const float* v       = (const float*)d_in[2];
    const int*   keylen  = (const int*)d_in[3];
    const int*   sampled = (const int*)d_in[4];
    const float* Wqkv    = (const float*)d_in[5];
    const float* bqkv    = (const float*)d_in[6];
    const float* Wo      = (const float*)d_in[7];
    const float* bo      = (const float*)d_in[8];
    float* out = (float*)d_out;

    char* ws = (char*)d_ws;
    size_t off = 0;
    auto take = [&](size_t bytes) -> char* {
        char* p = ws + off;
        off += (bytes + 255) & ~(size_t)255;
        return p;
    };
    bf16_t* Wqkv_b = (bf16_t*)take((size_t)3 * DD * DD * 2);
    bf16_t* Wo_b   = (bf16_t*)take((size_t)DD * DD * 2);
    bf16_t* qm     = (bf16_t*)take((size_t)MQ * DD * 2);
    bf16_t* kg     = (bf16_t*)take((size_t)SS * BB * DD * 2);
    bf16_t* vg     = (bf16_t*)take((size_t)SS * BB * DD * 2);
    bf16_t* qpm    = (bf16_t*)take((size_t)MQ * DD * 2);
    bf16_t* kpm    = (bf16_t*)take((size_t)SS * BB * DD * 2);
    bf16_t* vpm    = (bf16_t*)take((size_t)SS * BB * DD * 2);
    bf16_t* ctxm   = (bf16_t*)take((size_t)MQ * DD * 2);

    // conversions / gather
    {
        long long n4 = (long long)3 * DD * DD / 4;
        cvt_f32_bf16_kernel<<<(int)((n4 + 255) / 256), 256, 0, stream>>>(Wqkv, Wqkv_b, n4);
    }
    {
        long long n4 = (long long)DD * DD / 4;
        cvt_f32_bf16_kernel<<<(int)((n4 + 255) / 256), 256, 0, stream>>>(Wo, Wo_b, n4);
    }
    {
        long long n4 = (long long)MQ * DD / 4;
        cvt_f32_bf16_kernel<<<(int)((n4 + 255) / 256), 256, 0, stream>>>(q, qm, n4);
    }
    gather_cvt_kernel<<<SS * BB, 256, 0, stream>>>(k, v, sampled, kg, vg);

    // projections
    dim3 gq(MQ / 128, DD / 128);
    gemm_bf16_kernel<<<gq, 256, 0, stream>>>(qm, Wqkv_b, bqkv, qpm, nullptr,
                                             MQ, DD, DD, 0);
    dim3 gkv((SS * BB) / 128, DD / 128);
    gemm_bf16_kernel<<<gkv, 256, 0, stream>>>(kg, Wqkv_b + (size_t)DD * DD,
                                              bqkv + DD, kpm, nullptr,
                                              SS * BB, DD, DD, 0);
    gemm_bf16_kernel<<<gkv, 256, 0, stream>>>(vg, Wqkv_b + (size_t)2 * DD * DD,
                                              bqkv + 2 * DD, vpm, nullptr,
                                              SS * BB, DD, DD, 0);

    // attention
    dim3 ga(LQ / 64, BB * HH);
    attn_kernel<<<ga, 128, 0, stream>>>(qpm, kpm, vpm, sampled, keylen, ctxm);

    // output projection (f32 out + bias)
    gemm_bf16_kernel<<<gq, 256, 0, stream>>>(ctxm, Wo_b, bo, nullptr, out,
                                             MQ, DD, DD, 1);
}
